// MultiHeadCausalSelfAttention_71975061947020
// MI455X (gfx1250) — matrix operand order repaired and checked by reading, not compiled, not run on hardware
//
#include <hip/hip_runtime.h>

// ---------------------------------------------------------------------------
// MI455X (gfx1250) causal MHA forward.
// B=4, S=2048, E=1024, H=16, D=64.  All matmul stages use
// v_wmma_f32_16x16x32_bf16 (wave32 WMMA, f32 accumulate).
// Tile staging uses CDNA5 async global->LDS copies (ASYNCcnt).
// ---------------------------------------------------------------------------

typedef __attribute__((ext_vector_type(16))) __bf16        bf16x16;
typedef __attribute__((ext_vector_type(8)))  float         f32x8;
typedef __attribute__((ext_vector_type(4)))  unsigned int  u32x4;

union Frag {
  bf16x16        v;
  u32x4          q[2];
  unsigned short s[16];
};

__device__ __forceinline__ f32x8 zero8() {
  f32x8 z = {0.f, 0.f, 0.f, 0.f, 0.f, 0.f, 0.f, 0.f};
  return z;
}

// round-to-nearest-even f32 -> bf16 bits
__device__ __forceinline__ unsigned short f2bf(float f) {
  unsigned int u = __float_as_uint(f);
  u += 0x7FFFu + ((u >> 16) & 1u);
  return (unsigned short)(u >> 16);
}

// Generic pointers to __shared__ carry the LDS offset in the low 32 bits.
__device__ __forceinline__ unsigned int lds_addr(const void* p) {
  return (unsigned int)(unsigned long long)(uintptr_t)p;
}

// CDNA5 async copy: 16 bytes global -> LDS per active lane, ASYNCcnt-tracked.
__device__ __forceinline__ void async_b128(unsigned int lds, const void* g) {
  asm volatile("global_load_async_to_lds_b128 %0, %1, off"
               :: "v"(lds), "v"((unsigned long long)(uintptr_t)g)
               : "memory");
}
__device__ __forceinline__ void wait_async0() {
  asm volatile("s_wait_asynccnt 0x0" ::: "memory");
}

// ---------------------------------------------------------------------------
// fp32 -> bf16 conversion (grid-stride)
// ---------------------------------------------------------------------------
__global__ void cvt_f32_bf16(const float* __restrict__ src,
                             unsigned short* __restrict__ dst, int n) {
  for (int i = blockIdx.x * blockDim.x + threadIdx.x; i < n;
       i += gridDim.x * blockDim.x)
    dst[i] = f2bf(src[i]);
}

// ---------------------------------------------------------------------------
// GEMM1: qkv = x @ w_attn + b_attn, scattered into q/k/v [B,H,S,D] bf16.
// M=8192 N=3072 K=1024.  Block tile 128x64, BK=32, 8 waves (wave32).
// ---------------------------------------------------------------------------
__global__ __launch_bounds__(256)
void qkv_gemm_kernel(const unsigned short* __restrict__ A,   // [8192,1024] bf16
                     const unsigned short* __restrict__ Bw,  // [1024,3072] bf16
                     const float* __restrict__ bias,         // [3072]
                     unsigned short* __restrict__ Qo,
                     unsigned short* __restrict__ Ko,
                     unsigned short* __restrict__ Vo) {
  constexpr int K = 1024, N = 3072;
  constexpr int BM = 128, BN = 64, BK = 32;
  constexpr int LDA = BK + 8, LDB = BK + 8;
  __shared__ __align__(16) unsigned short As[BM * LDA];
  __shared__ __align__(16) unsigned short Bs[BN * LDB];  // transposed [n][k]

  const int tid = threadIdx.x, lane = tid & 31, w = tid >> 5;
  const int m0 = blockIdx.y * BM;
  const int n0 = blockIdx.x * BN;
  const int nfr = lane & 15, halfk = lane >> 4;

  f32x8 acc[4];
  for (int t = 0; t < 4; ++t) acc[t] = zero8();

  for (int k0 = 0; k0 < K; k0 += BK) {
    // A tile: async global->LDS, 16B chunks (4 chunks per 32-elem row)
    for (int i = tid; i < BM * BK / 8; i += 256) {
      int r = i >> 2, c8 = (i & 3) * 8;
      async_b128(lds_addr(&As[r * LDA + c8]),
                 &A[(size_t)(m0 + r) * K + k0 + c8]);
    }
    // B tile transposed: vector global read, scalar LDS scatter
    for (int i = tid; i < BK * BN / 8; i += 256) {
      int kk = i / (BN / 8), nc = (i % (BN / 8)) * 8;
      Frag tmp;
      tmp.q[0] = *(const u32x4*)&Bw[(size_t)(k0 + kk) * N + n0 + nc];
      for (int j = 0; j < 8; ++j) Bs[(nc + j) * LDB + kk] = tmp.s[j];
    }
    wait_async0();
    __syncthreads();

    Frag af;  // A frag: lanes 0-15 K 0-15, lanes 16-31 K 16-31
    {
      int row = 16 * w + nfr;
      int c = halfk * 16;
      af.q[0] = *(const u32x4*)&As[row * LDA + c];
      af.q[1] = *(const u32x4*)&As[row * LDA + c + 8];
    }
    Frag bfr[4];  // load all B frags first so DS waits batch up
    for (int t = 0; t < 4; ++t) {
      int nn = 16 * t + nfr;
      int c = halfk * 16;
      bfr[t].q[0] = *(const u32x4*)&Bs[nn * LDB + c];
      bfr[t].q[1] = *(const u32x4*)&Bs[nn * LDB + c + 8];
    }
    for (int t = 0; t < 4; ++t)
      acc[t] = __builtin_amdgcn_wmma_f32_16x16x32_bf16(
          false, af.v, false, bfr[t].v, (short)0, acc[t], false, false);
    __syncthreads();
  }

  // epilogue: +bias, fold 1/sqrt(D) into Q, scatter into [B,H,S,D]
  for (int t = 0; t < 4; ++t)
    for (int r = 0; r < 8; ++r) {
      int mg = m0 + 16 * w + r + 8 * halfk;
      int cg = n0 + 16 * t + nfr;
      float vv = acc[t][r] + bias[cg];
      int which = cg >> 10;        // 0=q 1=k 2=v
      int e = cg & 1023;
      int hh = e >> 6, dd = e & 63;
      int bb = mg >> 11, ss = mg & 2047;
      if (which == 0) vv *= 0.125f;  // 1/sqrt(64)
      unsigned short* dst = which == 0 ? Qo : (which == 1 ? Ko : Vo);
      dst[(((size_t)(bb * 16 + hh)) * 2048 + ss) * 64 + dd] = f2bf(vv);
    }
}

// ---------------------------------------------------------------------------
// Flash attention: one wave per (b, h, 16-query block). Online softmax.
// V tile is async-staged to LDS while QK^T + softmax run.
// ---------------------------------------------------------------------------
__global__ __launch_bounds__(32)
void flash_kernel(const unsigned short* __restrict__ Q,
                  const unsigned short* __restrict__ Kd,
                  const unsigned short* __restrict__ Vd,
                  unsigned short* __restrict__ Att) {
  constexpr int S = 2048, D = 64, H = 16, E = 1024;
  constexpr int LDP = 40;
  __shared__ __align__(16) unsigned short Pld[16 * LDP];
  __shared__ __align__(16) unsigned short Vs[32 * 64];  // [key][d] tile

  const int lane = threadIdx.x;
  const int nfr = lane & 15, half = lane >> 4;
  const int qb = blockIdx.x, hh = blockIdx.y, bb = blockIdx.z;
  const size_t base = ((size_t)(bb * H + hh)) * S * D;
  const unsigned short* Qp = Q + base;
  const unsigned short* Kp = Kd + base;
  const unsigned short* Vp = Vd + base;
  const int qrow0 = qb * 16;

  // Q fragments (K-dim = D = 64 -> two K=32 fragments), already scaled
  Frag qf[2];
  for (int f = 0; f < 2; ++f) {
    int row = qrow0 + nfr;
    int c = f * 32 + half * 16;
    qf[f].q[0] = *(const u32x4*)&Qp[(size_t)row * D + c];
    qf[f].q[1] = *(const u32x4*)&Qp[(size_t)row * D + c + 8];
  }

  float mrun[8], lrun[8];
  for (int r = 0; r < 8; ++r) { mrun[r] = -1e30f; lrun[r] = 0.f; }
  f32x8 acc[4];
  for (int t = 0; t < 4; ++t) acc[t] = zero8();

  for (int kb = 0; kb < qrow0 + 16; kb += 32) {
    // async-stage V tile [32 keys x 64 d] (overlaps with QK^T + softmax)
    for (int c = 0; c < 8; ++c)
      async_b128(lds_addr(&Vs[lane * 64 + c * 8]),
                 &Vp[(size_t)(kb + lane) * D + c * 8]);

    // S = Q K^T : Kp row-major [key][d] == B-matrix columns, contiguous loads
    f32x8 st[2];
    for (int t = 0; t < 2; ++t) {
      st[t] = zero8();
      for (int f = 0; f < 2; ++f) {
        Frag kf;
        int row = kb + 16 * t + nfr;
        int c = f * 32 + half * 16;
        kf.q[0] = *(const u32x4*)&Kp[(size_t)row * D + c];
        kf.q[1] = *(const u32x4*)&Kp[(size_t)row * D + c + 8];
        st[t] = __builtin_amdgcn_wmma_f32_16x16x32_bf16(
            false, qf[f].v, false, kf.v, (short)0, st[t], false, false);
      }
    }
    // causal mask + online softmax (row m = r + 8*half, col n = lane&15)
    for (int r = 0; r < 8; ++r) {
      int qi = qrow0 + r + 8 * half;
      float s0 = (kb + nfr <= qi) ? st[0][r] : -1e30f;
      float s1 = (kb + 16 + nfr <= qi) ? st[1][r] : -1e30f;
      float v = fmaxf(s0, s1);
      for (int off = 8; off; off >>= 1) v = fmaxf(v, __shfl_xor(v, off, 16));
      float mn = fmaxf(mrun[r], v);
      float corr = __expf(mrun[r] - mn);
      float p0 = __expf(s0 - mn);
      float p1 = __expf(s1 - mn);
      float rs = p0 + p1;
      for (int off = 8; off; off >>= 1) rs += __shfl_xor(rs, off, 16);
      lrun[r] = lrun[r] * corr + rs;
      mrun[r] = mn;
      for (int t = 0; t < 4; ++t) acc[t][r] *= corr;
      int m = r + 8 * half;
      Pld[m * LDP + nfr] = f2bf(p0);
      Pld[m * LDP + 16 + nfr] = f2bf(p1);
    }
    wait_async0();
    __syncthreads();  // C-layout -> A-fragment re-layout through LDS
    Frag pf;
    {
      int c = half * 16;
      pf.q[0] = *(const u32x4*)&Pld[nfr * LDP + c];
      pf.q[1] = *(const u32x4*)&Pld[nfr * LDP + c + 8];
    }
    // O += P V : per-lane B-matrix column gather from the staged LDS tile
    for (int t = 0; t < 4; ++t) {
      Frag vf;
      for (int e = 0; e < 16; ++e)
        vf.s[e] = Vs[(half * 16 + e) * 64 + 16 * t + nfr];
      acc[t] = __builtin_amdgcn_wmma_f32_16x16x32_bf16(
          false, pf.v, false, vf.v, (short)0, acc[t], false, false);
    }
    __syncthreads();
  }

  // normalize and store to att[B,S,E] bf16
  for (int t = 0; t < 4; ++t)
    for (int r = 0; r < 8; ++r) {
      int m = r + 8 * half;
      float o = acc[t][r] / lrun[r];
      Att[((size_t)bb * S + qrow0 + m) * E + hh * D + 16 * t + nfr] = f2bf(o);
    }
}

// ---------------------------------------------------------------------------
// GEMM2: out = att @ w_out + b_out   (M=8192 N=1024 K=1024, fp32 output)
// ---------------------------------------------------------------------------
__global__ __launch_bounds__(256)
void out_gemm_kernel(const unsigned short* __restrict__ A,   // [8192,1024] bf16
                     const unsigned short* __restrict__ Bw,  // [1024,1024] bf16
                     const float* __restrict__ bias,         // [1024]
                     float* __restrict__ Out) {
  constexpr int K = 1024, N = 1024;
  constexpr int BM = 128, BN = 64, BK = 32;
  constexpr int LDA = BK + 8, LDB = BK + 8;
  __shared__ __align__(16) unsigned short As[BM * LDA];
  __shared__ __align__(16) unsigned short Bs[BN * LDB];

  const int tid = threadIdx.x, lane = tid & 31, w = tid >> 5;
  const int m0 = blockIdx.y * BM;
  const int n0 = blockIdx.x * BN;
  const int nfr = lane & 15, halfk = lane >> 4;

  f32x8 acc[4];
  for (int t = 0; t < 4; ++t) acc[t] = zero8();

  for (int k0 = 0; k0 < K; k0 += BK) {
    for (int i = tid; i < BM * BK / 8; i += 256) {
      int r = i >> 2, c8 = (i & 3) * 8;
      async_b128(lds_addr(&As[r * LDA + c8]),
                 &A[(size_t)(m0 + r) * K + k0 + c8]);
    }
    for (int i = tid; i < BK * BN / 8; i += 256) {
      int kk = i / (BN / 8), nc = (i % (BN / 8)) * 8;
      Frag tmp;
      tmp.q[0] = *(const u32x4*)&Bw[(size_t)(k0 + kk) * N + n0 + nc];
      for (int j = 0; j < 8; ++j) Bs[(nc + j) * LDB + kk] = tmp.s[j];
    }
    wait_async0();
    __syncthreads();

    Frag af;
    {
      int row = 16 * w + nfr;
      int c = halfk * 16;
      af.q[0] = *(const u32x4*)&As[row * LDA + c];
      af.q[1] = *(const u32x4*)&As[row * LDA + c + 8];
    }
    Frag bfr[4];
    for (int t = 0; t < 4; ++t) {
      int nn = 16 * t + nfr;
      int c = halfk * 16;
      bfr[t].q[0] = *(const u32x4*)&Bs[nn * LDB + c];
      bfr[t].q[1] = *(const u32x4*)&Bs[nn * LDB + c + 8];
    }
    for (int t = 0; t < 4; ++t)
      acc[t] = __builtin_amdgcn_wmma_f32_16x16x32_bf16(
          false, af.v, false, bfr[t].v, (short)0, acc[t], false, false);
    __syncthreads();
  }

  for (int t = 0; t < 4; ++t)
    for (int r = 0; r < 8; ++r) {
      int mg = m0 + 16 * w + r + 8 * halfk;
      int cg = n0 + 16 * t + nfr;
      Out[(size_t)mg * N + cg] = acc[t][r] + bias[cg];
    }
}

// ---------------------------------------------------------------------------
// Host launcher
// ---------------------------------------------------------------------------
extern "C" void kernel_launch(void* const* d_in, const int* in_sizes, int n_in,
                              void* d_out, int out_size, void* d_ws,
                              size_t ws_size, hipStream_t stream) {
  (void)in_sizes; (void)n_in; (void)out_size; (void)ws_size;
  const float* x      = (const float*)d_in[0];
  const float* w_attn = (const float*)d_in[1];
  const float* b_attn = (const float*)d_in[2];
  const float* w_out  = (const float*)d_in[3];
  const float* b_out  = (const float*)d_in[4];
  float* out = (float*)d_out;

  // workspace carve (bf16 buffers), ~92.3 MB total
  char* ws = (char*)d_ws;
  unsigned short* xb  = (unsigned short*)(ws + 0);         // 8192x1024
  unsigned short* wab = (unsigned short*)(ws + 16777216);  // 1024x3072
  unsigned short* wob = (unsigned short*)(ws + 23068672);  // 1024x1024
  unsigned short* qb  = (unsigned short*)(ws + 25165824);  // [B,H,S,D]
  unsigned short* kb  = (unsigned short*)(ws + 41943040);
  unsigned short* vb  = (unsigned short*)(ws + 58720256);
  unsigned short* att = (unsigned short*)(ws + 75497472);  // [B,S,E]

  cvt_f32_bf16<<<2048, 256, 0, stream>>>(x, xb, 8388608);
  cvt_f32_bf16<<<2048, 256, 0, stream>>>(w_attn, wab, 3145728);
  cvt_f32_bf16<<<1024, 256, 0, stream>>>(w_out, wob, 1048576);

  qkv_gemm_kernel<<<dim3(48, 64), 256, 0, stream>>>(xb, wab, b_attn, qb, kb, vb);
  flash_kernel<<<dim3(128, 16, 4), 32, 0, stream>>>(qb, kb, vb, att);
  out_gemm_kernel<<<dim3(16, 64), 256, 0, stream>>>(att, wob, b_out, out);
}
